// GlimpseAgent_48052094108071
// MI455X (gfx1250) — matrix operand order repaired and checked by reading, not compiled, not run on hardware
//
#include <hip/hip_runtime.h>
#include <hip/hip_bf16.h>
#include <math.h>

// Problem constants (from reference)
#define S_IMG 224
#define PP    32
#define EE    512
#define NCLS_ 1000
#define BB    64
#define CC    3
#define TT    8
#define GAMMA_     0.96f
#define STEP_COST_ 0.001f
#define TAU_       0.5f

#define KCH 64   // K-chunk per barrier round (two WMMA k-steps); all GEMM K are multiples of 64

typedef __attribute__((ext_vector_type(16))) __bf16 v16bf;
typedef __attribute__((ext_vector_type(8)))  float  v8f;

// ---------------- helpers ----------------
__device__ __forceinline__ unsigned short f2bf(float f) {
  unsigned int u = __float_as_uint(f);
  u += 0x7FFFu + ((u >> 16) & 1u);          // round-to-nearest-even
  return (unsigned short)(u >> 16);
}
__device__ __forceinline__ float softplusf_(float x) {
  return fmaxf(x, 0.f) + log1pf(expf(-fabsf(x)));
}
__device__ __forceinline__ float u01(unsigned a, unsigned b, unsigned c) {
  unsigned h = a * 0x9E3779B1u + b * 0x85EBCA77u + c * 0xC2B2AE3Du + 0x27220A95u;
  h ^= h >> 16; h *= 0x7FEB352Du; h ^= h >> 15; h *= 0x846CA68Bu; h ^= h >> 16;
  return ((float)(h >> 8) + 0.5f) * (1.0f / 16777216.0f);   // (0,1)
}

// ---------------- bf16 WMMA GEMM: C[M,N] = act(A[M,K] * W + bias [+C]) ----------------
// W layout: trans==0 -> W[k*N+n] ; trans==1 -> W[n*K+k] (i.e. x @ W.T with W[N,K])
// Requirements (met by all call sites): K % 64 == 0, N % 4 == 0.
// Block: 128 threads = 4 wave32; each wave owns a 16-row tile; all share one 16-col slab.
__global__ __launch_bounds__(128)
void gemm_bf16_wmma(const float* __restrict__ A, const float* __restrict__ W,
                    const float* __restrict__ bias, float* __restrict__ C,
                    int M, int N, int K, int trans, int accum, int relu)
{
  // Row stride 68 ushorts = 136B: 8B-aligned rows (b64 LDS stores), 34-dword stride
  // (34*r mod 64 distinct over 16 rows -> conflict-free fragment gathers).
  __shared__ unsigned short Alds[64][KCH + 4];
  __shared__ unsigned short Blds[16][KCH + 4];
  const int tid  = threadIdx.x;
  const int lane = tid & 31;
  const int wv   = tid >> 5;
  const int n0   = blockIdx.x * 16;
  const int m0   = blockIdx.y * 64;

  v8f acc = {0.f, 0.f, 0.f, 0.f, 0.f, 0.f, 0.f, 0.f};

  for (int k0 = 0; k0 < K; k0 += KCH) {
    // ---- stage A tile: 64 rows x 64 K, float4 global loads, bf16-pack, b64 LDS stores ----
#pragma unroll
    for (int j = 0; j < 8; ++j) {
      int i  = tid + j * 128;            // 0..1023 float4 slots
      int r  = i >> 4;                   // 16 float4 per row
      int k  = (i & 15) * 4;
      int gm = m0 + r;
      float4 v = make_float4(0.f, 0.f, 0.f, 0.f);
      if (gm < M) v = *(const float4*)&A[(size_t)gm * K + (k0 + k)];
      uint2 p;
      p.x = (unsigned)f2bf(v.x) | ((unsigned)f2bf(v.y) << 16);
      p.y = (unsigned)f2bf(v.z) | ((unsigned)f2bf(v.w) << 16);
      *(uint2*)&Alds[r][k] = p;
    }
    // ---- stage B slab: 16 cols x 64 K ----
    if (trans) {                          // W[N,K]: K contiguous -> float4 along K
#pragma unroll
      for (int j = 0; j < 2; ++j) {
        int i   = tid + j * 128;          // 0..255 float4 slots
        int cix = i >> 4;                 // 16 float4 per column
        int k   = (i & 15) * 4;
        int gn  = n0 + cix;
        float4 v = make_float4(0.f, 0.f, 0.f, 0.f);
        if (gn < N) {
          v = *(const float4*)&W[(size_t)gn * K + (k0 + k)];
          if (k0 + KCH < K)
            __builtin_prefetch(&W[(size_t)gn * K + (k0 + KCH + k)], 0, 1);
        }
        uint2 p;
        p.x = (unsigned)f2bf(v.x) | ((unsigned)f2bf(v.y) << 16);
        p.y = (unsigned)f2bf(v.z) | ((unsigned)f2bf(v.w) << 16);
        *(uint2*)&Blds[cix][k] = p;
      }
    } else {                              // W[K,N]: N contiguous -> float4 along N
#pragma unroll
      for (int j = 0; j < 2; ++j) {
        int i  = tid + j * 128;           // 0..255: (kk, n-group)
        int kk = i >> 2;                  // 0..63
        int nn = (i & 3) * 4;             // 0,4,8,12
        int gn = n0 + nn;
        const float* src = &W[(size_t)(k0 + kk) * N + gn];
        float4 v;
        if (gn + 3 < N) {
          v = *(const float4*)src;        // N%4==0 keeps this 16B-aligned
          if (k0 + KCH < K)
            __builtin_prefetch(&W[(size_t)(k0 + KCH + kk) * N + gn], 0, 1);
        } else {
          v.x = (gn + 0 < N) ? src[0] : 0.f;
          v.y = (gn + 1 < N) ? src[1] : 0.f;
          v.z = (gn + 2 < N) ? src[2] : 0.f;
          v.w = (gn + 3 < N) ? src[3] : 0.f;
        }
        Blds[nn + 0][kk] = f2bf(v.x);
        Blds[nn + 1][kk] = f2bf(v.y);
        Blds[nn + 2][kk] = f2bf(v.z);
        Blds[nn + 3][kk] = f2bf(v.w);
      }
    }
    __syncthreads();

    // ---- two WMMAs per barrier round, A and B fragments in contiguous k order (k = 16h + s) ----
#pragma unroll
    for (int kc = 0; kc < KCH; kc += 32) {
      union { v16bf v; unsigned int u[8]; } fa, fb;
      const int mrow = wv * 16 + (lane & 15);
      const int ka = kc + ((lane < 16) ? 0 : 16);  // A: K 0..15 / 16..31
#pragma unroll
      for (int j = 0; j < 4; ++j) {
        fa.u[j]     = *(const unsigned int*)&Alds[mrow][ka + 2 * j];
        fa.u[4 + j] = *(const unsigned int*)&Alds[mrow][ka + 8 + 2 * j];
      }
      const int bc = lane & 15;                    // B: lane = column
      const int kb = kc + ((lane < 16) ? 0 : 16);  // K 0..15 / 16..31
#pragma unroll
      for (int j = 0; j < 8; ++j)
        fb.u[j] = *(const unsigned int*)&Blds[bc][kb + 2 * j];

      acc = __builtin_amdgcn_wmma_f32_16x16x32_bf16(false, fa.v, false, fb.v,
                                                    (short)0, acc, false, false);
    }
    __syncthreads();
  }

  const int col = n0 + (lane & 15);
  if (col < N) {
    const float bv = bias ? bias[col] : 0.f;
    const int mo = (lane < 16) ? 0 : 8;            // C/D layout: VGPR r -> M=r / 8+r
#pragma unroll
    for (int r = 0; r < 8; ++r) {
      int row = m0 + wv * 16 + mo + r;
      if (row < M) {
        size_t o = (size_t)row * N + col;
        float v = acc[r] + bv;
        if (accum) v += C[o];
        if (relu)  v = fmaxf(v, 0.f);
        C[o] = v;
      }
    }
  }
}

// ---------------- out[r] = dot(A[r,:K], w) + b0 ----------------
__global__ void matvec_rows(const float* __restrict__ A, const float* __restrict__ w,
                            const float* __restrict__ b0, float* __restrict__ out, int K)
{
  __shared__ float red[128];
  const int r = blockIdx.x, tid = threadIdx.x;
  float s = 0.f;
  for (int k = tid; k < K; k += 128) s += A[(size_t)r * K + k] * w[k];
  red[tid] = s; __syncthreads();
  for (int o = 64; o > 0; o >>= 1) { if (tid < o) red[tid] += red[tid + o]; __syncthreads(); }
  if (tid == 0) out[r] = red[0] + b0[0];
}

// ---------------- Gumbel-max categorical + log_softmax + entropy over a_logits row ----------------
__global__ void policy_sample(const float* __restrict__ logits, int NS, int t,
                              float* __restrict__ lp_out, float* __restrict__ ent_out,
                              int* __restrict__ idx_out)
{
  __shared__ float sA[256], sB[256], sC[256];
  __shared__ int   sI[256];
  __shared__ float shGmax, shChos; __shared__ int shIdx;
  const int b = blockIdx.x, tid = threadIdx.x;
  const float* row = logits + (size_t)b * NS;

  float lmax = -3.4e38f, pmax = -3.4e38f, plog = 0.f; int pidx = 0;
  for (int j = tid; j < NS; j += 256) {
    float l = row[j];
    float u = u01(0xA5117u + (unsigned)t, (unsigned)b, (unsigned)j);
    float p = l - logf(-logf(u));                        // Gumbel perturbation
    lmax = fmaxf(lmax, l);
    if (p > pmax) { pmax = p; pidx = j; plog = l; }
  }
  sA[tid] = lmax; sB[tid] = pmax; sC[tid] = plog; sI[tid] = pidx;
  __syncthreads();
  for (int o = 128; o > 0; o >>= 1) {
    if (tid < o) {
      sA[tid] = fmaxf(sA[tid], sA[tid + o]);
      if (sB[tid + o] > sB[tid]) { sB[tid] = sB[tid + o]; sC[tid] = sC[tid + o]; sI[tid] = sI[tid + o]; }
    }
    __syncthreads();
  }
  if (tid == 0) { shGmax = sA[0]; shChos = sC[0]; shIdx = sI[0]; }
  __syncthreads();
  const float gmax = shGmax;
  float se = 0.f, sle = 0.f;
  for (int j = tid; j < NS; j += 256) {
    float l = row[j] - gmax;
    float e = expf(l);
    se += e; sle += l * e;
  }
  __syncthreads();
  sA[tid] = se; sB[tid] = sle; __syncthreads();
  for (int o = 128; o > 0; o >>= 1) { if (tid < o) { sA[tid] += sA[tid + o]; sB[tid] += sB[tid + o]; } __syncthreads(); }
  if (tid == 0) {
    float logZ = logf(sA[0]);
    lp_out[t * BB + b]  = shChos - gmax - logZ;          // lp_a (lp_s added later)
    ent_out[t * BB + b] = logZ - sB[0] / sA[0];          // ent_a (ent_s added later)
    idx_out[b] = shIdx;
  }
}

// ---------------- stop head: sample Bernoulli, fold lp_s / ent_s into outputs ----------------
__global__ void stop_update(const float* __restrict__ z, int t,
                            float* __restrict__ lp_out, float* __restrict__ ent_out)
{
  const int b = threadIdx.x; if (b >= BB) return;
  float zv = z[b];
  float pstop = 1.f / (1.f + expf(-zv));
  float u = u01(0xB0B0u + (unsigned)t, (unsigned)b, 0x5151u);
  float stop = (u < pstop) ? 1.f : 0.f;
  float spn = softplusf_(-zv), spp = softplusf_(zv);
  lp_out[t * BB + b]  += -(stop * spn + (1.f - stop) * spp);
  ent_out[t * BB + b] +=  pstop * spn + (1.f - pstop) * spp;
}

// ---------------- bilinear glimpse patch extraction (zero padding) ----------------
__global__ void glimpse_patch(const float* __restrict__ x, const int* __restrict__ idx,
                              float* __restrict__ patch)
{
  int gid = blockIdx.x * blockDim.x + threadIdx.x;
  if (gid >= BB * CC * PP * PP) return;
  int px = gid & (PP - 1);
  int py = (gid >> 5) & (PP - 1);
  int c  = (gid >> 10) % CC;
  int b  = gid / (CC * PP * PP);
  int id = idx[b];
  float colf = (float)(id % S_IMG);
  float rowf = (float)(id / S_IMG);
  float xx = colf + ((float)px - (PP - 1) * 0.5f);
  float yy = rowf + ((float)py - (PP - 1) * 0.5f);
  float x0 = floorf(xx), y0 = floorf(yy);
  float wx1 = xx - x0, wx0 = 1.f - wx1;
  float wy1 = yy - y0, wy0 = 1.f - wy1;
  const float* img = x + ((size_t)b * CC + c) * S_IMG * S_IMG;
  float acc = 0.f;
#pragma unroll
  for (int dy = 0; dy < 2; ++dy)
#pragma unroll
    for (int dx = 0; dx < 2; ++dx) {
      float yi = y0 + (float)dy, xi = x0 + (float)dx;
      float wgt = (dy ? wy1 : wy0) * (dx ? wx1 : wx0);
      bool valid = (yi >= 0.f) && (yi <= (float)(S_IMG - 1)) &&
                   (xi >= 0.f) && (xi <= (float)(S_IMG - 1));
      int yc = (int)yi; yc = yc < 0 ? 0 : (yc > S_IMG - 1 ? S_IMG - 1 : yc);
      int xc = (int)xi; xc = xc < 0 ? 0 : (xc > S_IMG - 1 ? S_IMG - 1 : xc);
      acc += wgt * (valid ? img[yc * S_IMG + xc] : 0.f);
    }
  patch[(size_t)b * (CC * PP * PP) + c * PP * PP + py * PP + px] = acc;
}

// ---------------- LSTM cell pointwise (gates in torch i,f,g,o order) ----------------
__global__ void lstm_pointwise(const float* __restrict__ g, float* __restrict__ c,
                               float* __restrict__ h, float* __restrict__ hout)
{
  int i = blockIdx.x * blockDim.x + threadIdx.x;
  if (i >= BB * EE) return;
  int b = i / EE, e = i % EE;
  const float* gr = g + (size_t)b * 4 * EE;
  float ig = 1.f / (1.f + expf(-gr[e]));
  float fg = 1.f / (1.f + expf(-gr[EE + e]));
  float gg = tanhf(gr[2 * EE + e]);
  float og = 1.f / (1.f + expf(-gr[3 * EE + e]));
  float cn = fg * c[i] + ig * gg;
  c[i] = cn;
  float hn = og * tanhf(cn);
  h[i] = hn;
  if (hout) hout[i] = hn;
}

// ---------------- temporal softmax attention pool over rnn_out ----------------
__global__ void attn_pool(const float* __restrict__ rnn_out, const float* __restrict__ scores,
                          float* __restrict__ pooled, int Tc)
{
  const int b = blockIdx.x, tid = threadIdx.x;
  float a[TT]; float m = -3.4e38f;
  for (int s = 0; s < Tc; ++s) { a[s] = scores[s * BB + b] / TAU_; m = fmaxf(m, a[s]); }
  float se = 0.f;
  for (int s = 0; s < Tc; ++s) { a[s] = expf(a[s] - m); se += a[s]; }
  for (int s = 0; s < Tc; ++s) a[s] /= se;
  for (int e = tid; e < EE; e += blockDim.x) {
    float acc = 0.f;
    for (int s = 0; s < Tc; ++s) acc += a[s] * rnn_out[((size_t)s * BB + b) * EE + e];
    pooled[(size_t)b * EE + e] = acc;
  }
}

// ---------------- confidence reward: tanh(CE_new - CE_old) - step cost ----------------
__global__ void ce_reward(const float* __restrict__ fl, float* __restrict__ prev,
                          const int* __restrict__ targets, float* __restrict__ rewards, int t)
{
  __shared__ float red[256];
  __shared__ float shm, shce_new, shce_old;
  const int b = blockIdx.x, tid = threadIdx.x;
  const float* rowN = fl + (size_t)b * NCLS_;
  float* rowP = prev + (size_t)b * NCLS_;
  const int tg = targets[b];

  float m = -3.4e38f;
  for (int j = tid; j < NCLS_; j += 256) m = fmaxf(m, rowN[j]);
  red[tid] = m; __syncthreads();
  for (int o = 128; o > 0; o >>= 1) { if (tid < o) red[tid] = fmaxf(red[tid], red[tid + o]); __syncthreads(); }
  if (!tid) shm = red[0];
  __syncthreads();
  float mm = shm, se = 0.f;
  for (int j = tid; j < NCLS_; j += 256) se += expf(rowN[j] - mm);
  __syncthreads(); red[tid] = se; __syncthreads();
  for (int o = 128; o > 0; o >>= 1) { if (tid < o) red[tid] += red[tid + o]; __syncthreads(); }
  if (!tid) shce_new = -(rowN[tg] - mm - logf(red[0]));
  __syncthreads();

  if (t > 0) {
    float m2 = -3.4e38f;
    for (int j = tid; j < NCLS_; j += 256) m2 = fmaxf(m2, rowP[j]);
    red[tid] = m2; __syncthreads();
    for (int o = 128; o > 0; o >>= 1) { if (tid < o) red[tid] = fmaxf(red[tid], red[tid + o]); __syncthreads(); }
    if (!tid) shm = red[0];
    __syncthreads();
    float mo = shm, se2 = 0.f;
    for (int j = tid; j < NCLS_; j += 256) se2 += expf(rowP[j] - mo);
    __syncthreads(); red[tid] = se2; __syncthreads();
    for (int o = 128; o > 0; o >>= 1) { if (tid < o) red[tid] += red[tid + o]; __syncthreads(); }
    if (!tid) shce_old = -(rowP[tg] - mo - logf(red[0]));
    __syncthreads();
  }
  for (int j = tid; j < NCLS_; j += 256) rowP[j] = rowN[j];   // prev_logits <- fl
  if (!tid)
    rewards[t * BB + b] = (t > 0) ? (tanhf(shce_new - shce_old) - STEP_COST_) : (-STEP_COST_);
}

// ---------------- final reward + discounted returns + advantages ----------------
__global__ void finalize_k(const float* __restrict__ logits, const int* __restrict__ targets,
                           float* __restrict__ rewards, const float* __restrict__ values,
                           float* __restrict__ adv, float* __restrict__ ret)
{
  const int b = threadIdx.x; if (b >= BB) return;
  const float* row = logits + (size_t)b * NCLS_;
  int am = 0; float mv = row[0];
  for (int j = 1; j < NCLS_; ++j) if (row[j] > mv) { mv = row[j]; am = j; }
  rewards[(TT - 1) * BB + b] += (am == targets[b]) ? 1.f : -1.f;
  float R = 0.f;
  for (int t = TT - 1; t >= 0; --t) {
    R = rewards[t * BB + b] + GAMMA_ * R;
    ret[t * BB + b] = R;
    adv[t * BB + b] = R - values[t * BB + b];
  }
}

__global__ void fill0(float* p, int n) {
  int i = blockIdx.x * blockDim.x + threadIdx.x;
  if (i < n) p[i] = 0.f;
}

// ---------------- host orchestration ----------------
extern "C" void kernel_launch(void* const* d_in, const int* in_sizes, int n_in,
                              void* d_out, int out_size, void* d_ws, size_t ws_size,
                              hipStream_t stream)
{
  (void)in_sizes; (void)n_in; (void)out_size; (void)ws_size;
  const float* x          = (const float*)d_in[0];
  const int*   targets    = (const int*)  d_in[1];
  const float* enc_W      = (const float*)d_in[2];
  const float* enc_b      = (const float*)d_in[3];
  const float* pol_W1     = (const float*)d_in[4];
  const float* pol_b1     = (const float*)d_in[5];
  const float* pol_dist_W = (const float*)d_in[6];
  const float* pol_dist_b = (const float*)d_in[7];
  const float* pol_stop_W = (const float*)d_in[8];
  const float* pol_stop_b = (const float*)d_in[9];
  const float* mem_Wih    = (const float*)d_in[10];
  const float* mem_Whh    = (const float*)d_in[11];
  const float* mem_bih    = (const float*)d_in[12];
  const float* mem_bhh    = (const float*)d_in[13];
  const float* rnn_Wih    = (const float*)d_in[14];
  const float* rnn_Whh    = (const float*)d_in[15];
  const float* rnn_bih    = (const float*)d_in[16];
  const float* rnn_bhh    = (const float*)d_in[17];
  const float* gate_W     = (const float*)d_in[18];
  const float* gate_b     = (const float*)d_in[19];
  const float* cls_W      = (const float*)d_in[20];
  const float* cls_b      = (const float*)d_in[21];
  const float* val_W      = (const float*)d_in[22];
  const float* val_b      = (const float*)d_in[23];
  // d_in[24] = max_steps (device scalar); reference setup uses 8 -> TT.

  float* out = (float*)d_out;
  float* logits_out = out;                  // [B,NCLS]  64000
  float* logps_out  = out + 64000;          // [T,B]     512
  float* adv_out    = out + 64512;          // [T,B]
  float* ret_out    = out + 65024;          // [T,B]
  float* vals_out   = out + 65536;          // [T,B]
  float* ents_out   = out + 66048;          // [T,B]

  char* ws = (char*)d_ws;
  size_t cur = 0;
  auto alloc = [&](size_t nfloats) -> float* {
    float* p = (float*)(ws + cur);
    cur += ((nfloats * sizeof(float) + 255) / 256) * 256;
    return p;
  };
  float* h_mem   = alloc((size_t)BB * EE);
  float* c_mem   = alloc((size_t)BB * EE);
  float* hid     = alloc((size_t)BB * EE);
  float* zbuf    = alloc(BB);
  int*   idxbuf  = (int*)alloc(BB);
  float* patch   = alloc((size_t)BB * CC * PP * PP);
  float* feats   = alloc((size_t)TT * BB * EE);
  float* gates   = alloc((size_t)BB * 4 * EE);
  float* h_rnn   = alloc((size_t)BB * EE);
  float* c_rnn   = alloc((size_t)BB * EE);
  float* rnn_out = alloc((size_t)TT * BB * EE);
  float* scores  = alloc((size_t)TT * BB);
  float* pooled  = alloc((size_t)BB * EE);
  float* fl      = alloc((size_t)BB * NCLS_);
  float* prevlog = alloc((size_t)BB * NCLS_);
  float* rewards = alloc((size_t)TT * BB);
  float* a_logit = alloc((size_t)BB * S_IMG * S_IMG);   // 12.85 MB, allocated last

  auto gemm = [&](const float* A, const float* W, const float* bias, float* Cc,
                  int M, int N, int K, int trans, int accum, int relu) {
    dim3 g((unsigned)((N + 15) / 16), (unsigned)((M + 63) / 64));
    gemm_bf16_wmma<<<g, 128, 0, stream>>>(A, W, bias, Cc, M, N, K, trans, accum, relu);
  };
  const int NE = BB * EE;

  fill0<<<(NE + 255) / 256, 256, 0, stream>>>(h_mem, NE);
  fill0<<<(NE + 255) / 256, 256, 0, stream>>>(c_mem, NE);

  for (int t = 0; t < TT; ++t) {
    // ---- policy on detached context (h_mem == prev_ctx) ----
    gemm(h_mem, pol_W1, pol_b1, hid, BB, EE, EE, 0, 0, 1);                         // relu
    gemm(hid, pol_dist_W, pol_dist_b, a_logit, BB, S_IMG * S_IMG, EE, 0, 0, 0);    // big stream
    matvec_rows<<<BB, 128, 0, stream>>>(hid, pol_stop_W, pol_stop_b, zbuf, EE);
    policy_sample<<<BB, 256, 0, stream>>>(a_logit, S_IMG * S_IMG, t, logps_out, ents_out, idxbuf);
    stop_update<<<1, BB, 0, stream>>>(zbuf, t, logps_out, ents_out);
    // ---- glimpse + encoder ----
    glimpse_patch<<<(BB * CC * PP * PP + 255) / 256, 256, 0, stream>>>(x, idxbuf, patch);
    float* feat_t = feats + (size_t)t * NE;
    gemm(patch, enc_W, enc_b, feat_t, BB, EE, CC * PP * PP, 0, 0, 0);
    // ---- value head on prev_ctx ----
    matvec_rows<<<BB, 128, 0, stream>>>(h_mem, val_W, val_b, vals_out + t * BB, EE);
    // ---- memory LSTM cell ----
    gemm(feat_t, mem_Wih, mem_bih, gates, BB, 4 * EE, EE, 1, 0, 0);
    gemm(h_mem,  mem_Whh, mem_bhh, gates, BB, 4 * EE, EE, 1, 1, 0);
    lstm_pointwise<<<(NE + 255) / 256, 256, 0, stream>>>(gates, c_mem, h_mem, nullptr);
    // ---- re-forward over growing sequence ----
    fill0<<<(NE + 255) / 256, 256, 0, stream>>>(h_rnn, NE);
    fill0<<<(NE + 255) / 256, 256, 0, stream>>>(c_rnn, NE);
    for (int s = 0; s <= t; ++s) {
      const float* fs = feats + (size_t)s * NE;
      gemm(fs,    rnn_Wih, rnn_bih, gates, BB, 4 * EE, EE, 1, 0, 0);
      gemm(h_rnn, rnn_Whh, rnn_bhh, gates, BB, 4 * EE, EE, 1, 1, 0);
      lstm_pointwise<<<(NE + 255) / 256, 256, 0, stream>>>(gates, c_rnn, h_rnn,
                                                           rnn_out + (size_t)s * NE);
    }
    matvec_rows<<<(t + 1) * BB, 128, 0, stream>>>(rnn_out, gate_W, gate_b, scores, EE);
    attn_pool<<<BB, 128, 0, stream>>>(rnn_out, scores, pooled, t + 1);
    gemm(pooled, cls_W, cls_b, fl, BB, NCLS_, EE, 0, 0, 0);
    ce_reward<<<BB, 256, 0, stream>>>(fl, prevlog, targets, rewards, t);
  }

  // Final logits == re-forward over all TT feats == fl from t=TT-1 (deterministic).
  (void)hipMemcpyAsync(logits_out, fl, (size_t)BB * NCLS_ * sizeof(float),
                       hipMemcpyDeviceToDevice, stream);
  finalize_k<<<1, BB, 0, stream>>>(logits_out, targets, rewards, vals_out, adv_out, ret_out);
}
